// Cross_Packing_Attention_3d_68444598829445
// MI455X (gfx1250) — compile-verified, matrix operand order checked by reading, not run on hardware
//
#include <hip/hip_runtime.h>

// ---------------- problem constants ----------------
#define VOL   (60*36*60)   // 129600 voxels, dims (D=60, H=36, W=60)
#define CP    3456         // space_to_channel channels = 6^3 * 16
#define MRAW  600          // packed spatial = 10*6*10
#define MPAD  608          // pad K to multiple of 32 for WMMA (19*32)

#define KC16  448          // conv K for Cin=16: 28 taps (27 real + 1 pad) * 16
#define KST16 464          // LDS row stride (halves): +16 pad, keeps 16B alignment
#define KC1   1728         // conv1 K: 27 taps * 64 virtual channels (exact, 54*32)
#define KST1  1744         // LDS row stride for conv1

// ---------------- WMMA vector types ----------------
typedef __attribute__((ext_vector_type(16))) __bf16 v16bf;
typedef __attribute__((ext_vector_type(8)))  __bf16 v8bf;
typedef __attribute__((ext_vector_type(8)))  float  v8f;

union V16 { v16bf v; v8bf h[2]; __bf16 e[16]; };
union F16 { float4 q[4]; float f[16]; };

// =====================================================================
// Weight repack: w[Cout][Cin][27] f32  ->  Wb[co][tap*Cin+cin] bf16,
// tap padded to KtapPad (zeros), co padded to CoutPad (zeros).
// Gives A-fragments the same contiguous-run layout as the GEMM kernels.
// =====================================================================
__global__ void repack_w(const float* __restrict__ w, __bf16* __restrict__ Wb,
                         int CoutReal, int Cin, int KtapPad, int total) {
  int idx = blockIdx.x * blockDim.x + threadIdx.x;
  if (idx >= total) return;
  int K = KtapPad * Cin;
  int co = idx / K, k = idx % K;
  int tap = k / Cin, cin = k % Cin;
  float v = 0.f;
  if (co < CoutReal && tap < 27)
    v = w[((size_t)co * Cin + cin) * 27 + tap];
  Wb[idx] = (__bf16)v;
}

// =====================================================================
// conv1 (WMMA implicit GEMM): ref_in = concat([x, flipD, flipH, flipW])
// (64 virtual ch) -> 16 ch, 3x3x3 SAME, +bias, ReLU.
// One wave per (z,y,x-tile): im2col B (1728 x 16) staged in LDS, 54 WMMA.
// =====================================================================
__global__ void conv1_wmma(const float* __restrict__ x, const __bf16* __restrict__ Wb,
                           const float* __restrict__ bias, float* __restrict__ out) {
  __shared__ __attribute__((aligned(16))) __bf16 Bt[16 * KST1];   // 55.8 KB
  const int lane = threadIdx.x;
  const int n = lane & 15, h = lane >> 4;
  const int zy = blockIdx.x;
  const int z = zy / 36, y = zy % 36;
  const int x0 = (blockIdx.y < 3) ? (int)blockIdx.y * 16 : 44;   // 0,16,32,44

  // ---- im2col fill (coalesced: 16 consecutive lanes -> 16 consecutive x) ----
  for (int idx = lane; idx < KC1 * 16; idx += 32) {
    int k = idx >> 4, nn = idx & 15;
    int tap = k / 64, cin = k & 63;
    int g = cin >> 4, c = cin & 15;
    int zz = z + tap / 9 - 1;
    int yy = y + (tap / 3) % 3 - 1;
    int xq = x0 + nn + tap % 3 - 1;
    float v = 0.f;
    if (zz >= 0 && zz < 60 && yy >= 0 && yy < 36 && xq >= 0 && xq < 60) {
      int rz = (g == 1) ? 59 - zz : zz;     // flip axis 2 (D)
      int ry = (g == 2) ? 35 - yy : yy;     // flip axis 3 (H)
      int rx = (g == 3) ? 59 - xq : xq;     // flip axis 4 (W)
      v = x[(size_t)c * VOL + (rz * 36 + ry) * 60 + rx];
    }
    Bt[nn * KST1 + k] = (__bf16)v;
  }
  // single wave: LDS ops are in-order within a wave, no barrier needed

  v8f acc = {};
  const __bf16* arow = Wb + (size_t)n * KC1;
  for (int k = 0; k < KC1; k += 32) {
    V16 a, b;
    a.h[0] = *(const v8bf*)(arow + k + 8 * h);
    a.h[1] = *(const v8bf*)(arow + k + 16 + 8 * h);
    b.h[0] = *(const v8bf*)&Bt[n * KST1 + k + 8 * h];
    b.h[1] = *(const v8bf*)&Bt[n * KST1 + k + 16 + 8 * h];
    acc = __builtin_amdgcn_wmma_f32_16x16x32_bf16(
        false, a.v, false, b.v, (short)0, acc, false, false);
  }
  int pos = (z * 36 + y) * 60 + x0;
  #pragma unroll
  for (int r = 0; r < 8; ++r) {
    int co = r + 8 * h;
    out[(size_t)co * VOL + pos + n] = fmaxf(acc[r] + bias[co], 0.f);
  }
}

// =====================================================================
// Generic Cin=16 conv (WMMA implicit GEMM), 3x3x3 SAME, +bias.
// COT = number of 16-row Cout tiles (1 for 16ch, 6 for 81ch padded to 96).
// One wave per (z,y,x-tile); B tile (448 x 16) staged once in LDS,
// reused by all COT output tiles.
// =====================================================================
template<int COT, bool RELU>
__global__ void conv_wmma_c16(const float* __restrict__ in, const __bf16* __restrict__ Wb,
                              const float* __restrict__ bias, float* __restrict__ out,
                              int CoutReal) {
  __shared__ __attribute__((aligned(16))) __bf16 Bt[16 * KST16];  // 14.8 KB
  const int lane = threadIdx.x;
  const int n = lane & 15, h = lane >> 4;
  const int zy = blockIdx.x;
  const int z = zy / 36, y = zy % 36;
  const int x0 = (blockIdx.y < 3) ? (int)blockIdx.y * 16 : 44;

  for (int idx = lane; idx < KC16 * 16; idx += 32) {
    int k = idx >> 4, nn = idx & 15;
    int tap = k >> 4, cin = k & 15;
    float v = 0.f;
    if (tap < 27) {
      int zz = z + tap / 9 - 1;
      int yy = y + (tap / 3) % 3 - 1;
      int xq = x0 + nn + tap % 3 - 1;
      if (zz >= 0 && zz < 60 && yy >= 0 && yy < 36 && xq >= 0 && xq < 60)
        v = in[(size_t)cin * VOL + (zz * 36 + yy) * 60 + xq];
    }
    Bt[nn * KST16 + k] = (__bf16)v;
  }

  v8f acc[COT] = {};
  for (int k = 0; k < KC16; k += 32) {
    V16 b;
    b.h[0] = *(const v8bf*)&Bt[n * KST16 + k + 8 * h];
    b.h[1] = *(const v8bf*)&Bt[n * KST16 + k + 16 + 8 * h];
    #pragma unroll
    for (int ct = 0; ct < COT; ++ct) {
      const __bf16* arow = Wb + (size_t)(ct * 16 + n) * KC16;
      V16 a;
      a.h[0] = *(const v8bf*)(arow + k + 8 * h);
      a.h[1] = *(const v8bf*)(arow + k + 16 + 8 * h);
      acc[ct] = __builtin_amdgcn_wmma_f32_16x16x32_bf16(
          false, a.v, false, b.v, (short)0, acc[ct], false, false);
    }
  }
  int pos = (z * 36 + y) * 60 + x0;
  #pragma unroll
  for (int ct = 0; ct < COT; ++ct)
    #pragma unroll
    for (int r = 0; r < 8; ++r) {
      int co = ct * 16 + r + 8 * h;
      if (co < CoutReal) {
        float v = acc[ct][r] + bias[co];
        if (RELU) v = fmaxf(v, 0.f);
        out[(size_t)co * VOL + pos + n] = v;
      }
    }
}

// =====================================================================
// space_to_channel(bs=6): fm[16][60][36][60] -> XS f32 [3456][600],
// QB bf16 [3456][608] (K-padded rows), QT bf16 [608][3456] (transpose).
//   c' = S2*96 + (S3%6)*16 + C ;  m = (S1/6)*60 + (S1%6)*10 + (S3/6)
// =====================================================================
__global__ void s2c_kernel(const float* __restrict__ fm, float* __restrict__ XS,
                           __bf16* __restrict__ QB, __bf16* __restrict__ QT) {
  int idx = blockIdx.x * blockDim.x + threadIdx.x;
  if (idx >= CP * MPAD) return;
  int cp = idx / MPAD, m = idx % MPAD;
  if (m < MRAW) {
    int w1 = m / 60, rem = m % 60, w2 = rem / 10, w3 = rem % 10;
    int S1 = w1 * 6 + w2;
    int S2 = cp / 96;
    int S3 = w3 * 6 + (cp / 16) % 6;
    int C  = cp % 16;
    float v = fm[((C * 60 + S1) * 36 + S2) * 60 + S3];
    XS[(size_t)cp * MRAW + m] = v;
    QB[(size_t)cp * MPAD + m] = (__bf16)v;
    QT[(size_t)m * CP + cp]   = (__bf16)v;
  } else {
    QB[(size_t)cp * MPAD + m] = (__bf16)0.f;
    QT[(size_t)m * CP + cp]   = (__bf16)0.f;
  }
}

// =====================================================================
// Energy GEMM:  E[c][d] = sum_m Q[c][m] * Q[d][m]   (3456x3456, K=608)
// Wave computes a 16x64 strip (A fragment reused 4x).
// =====================================================================
__global__ void energy_wmma(const __bf16* __restrict__ Qb, float* __restrict__ E) {
  const int lane = threadIdx.x & 31;
  const int wave = threadIdx.x >> 5;
  const int n = lane & 15;
  const int h = lane >> 4;
  const int c0 = blockIdx.x * 16;
  const int d0 = (blockIdx.y * 2 + wave) * 64;
  v8f acc[4] = {};
  const __bf16* arow = Qb + (size_t)(c0 + n) * MPAD;
  for (int k = 0; k < MPAD; k += 32) {
    V16 a;
    a.h[0] = *(const v8bf*)(arow + k + 8*h);
    a.h[1] = *(const v8bf*)(arow + k + 16 + 8*h);
    #pragma unroll
    for (int t = 0; t < 4; ++t) {
      const __bf16* brow = Qb + (size_t)(d0 + 16*t + n) * MPAD;
      V16 b;
      b.h[0] = *(const v8bf*)(brow + k + 8*h);
      b.h[1] = *(const v8bf*)(brow + k + 16 + 8*h);
      acc[t] = __builtin_amdgcn_wmma_f32_16x16x32_bf16(
          false, a.v, false, b.v, (short)0, acc[t], false, false);
    }
  }
  #pragma unroll
  for (int t = 0; t < 4; ++t)
    #pragma unroll
    for (int r = 0; r < 8; ++r)
      E[(size_t)(c0 + r + 8*h) * CP + d0 + 16*t + n] = acc[t][r];
}

// =====================================================================
// softmax(max_row(E) - E) over last axis == softmin(E). In-place, 1 block/row.
// =====================================================================
__global__ void softmax_row(float* __restrict__ E) {
  const int row = blockIdx.x;
  float* e = E + (size_t)row * CP;
  __shared__ float red[256];
  int tid = threadIdx.x;
  float mn = 3.4e38f;
  for (int i = tid; i < CP; i += 256) mn = fminf(mn, e[i]);
  red[tid] = mn; __syncthreads();
  for (int s = 128; s > 0; s >>= 1) {
    if (tid < s) red[tid] = fminf(red[tid], red[tid + s]);
    __syncthreads();
  }
  mn = red[0]; __syncthreads();
  float sum = 0.f;
  for (int i = tid; i < CP; i += 256) { float v = __expf(mn - e[i]); e[i] = v; sum += v; }
  red[tid] = sum; __syncthreads();
  for (int s = 128; s > 0; s >>= 1) {
    if (tid < s) red[tid] += red[tid + s];
    __syncthreads();
  }
  float inv = 1.f / red[0];
  for (int i = tid; i < CP; i += 256) e[i] *= inv;
}

// =====================================================================
// Output GEMM: O[c][m] = sum_d attn[c][d] * Q[d][m]  (3456x608, K=3456)
// A (attn) converted f32->bf16 in registers; B from QT.
// =====================================================================
__global__ void outmm_wmma(const float* __restrict__ A, const __bf16* __restrict__ Qt,
                           float* __restrict__ O) {
  const int lane = threadIdx.x & 31;
  const int n = lane & 15;
  const int h = lane >> 4;
  const int c0 = blockIdx.x * 16;
  const int m0 = blockIdx.y * 32;
  v8f acc[2] = {};
  const float*  arow = A  + (size_t)(c0 + n) * CP;
  const __bf16* br0  = Qt + (size_t)(m0 + n) * CP;
  const __bf16* br1  = Qt + (size_t)(m0 + 16 + n) * CP;
  for (int k = 0; k < CP; k += 32) {
    __builtin_prefetch(arow + k + 128, 0, 1);   // global_prefetch_b8
    F16 fa;
    fa.q[0] = *(const float4*)(arow + k + 8*h);
    fa.q[1] = *(const float4*)(arow + k + 8*h + 4);
    fa.q[2] = *(const float4*)(arow + k + 16 + 8*h);
    fa.q[3] = *(const float4*)(arow + k + 16 + 8*h + 4);
    V16 a;
    #pragma unroll
    for (int i = 0; i < 16; ++i) a.e[i] = (__bf16)fa.f[i];
    V16 b0, b1;
    b0.h[0] = *(const v8bf*)(br0 + k + 8*h);
    b0.h[1] = *(const v8bf*)(br0 + k + 16 + 8*h);
    b1.h[0] = *(const v8bf*)(br1 + k + 8*h);
    b1.h[1] = *(const v8bf*)(br1 + k + 16 + 8*h);
    acc[0] = __builtin_amdgcn_wmma_f32_16x16x32_bf16(
        false, a.v, false, b0.v, (short)0, acc[0], false, false);
    acc[1] = __builtin_amdgcn_wmma_f32_16x16x32_bf16(
        false, a.v, false, b1.v, (short)0, acc[1], false, false);
  }
  #pragma unroll
  for (int t = 0; t < 2; ++t)
    #pragma unroll
    for (int r = 0; r < 8; ++r)
      O[(size_t)(c0 + r + 8*h) * MPAD + m0 + 16*t + n] = acc[t][r];
}

// =====================================================================
// channel_to_space(gamma*O + XS) + ReLU -> feature map [16][60][36][60].
//   Cy = ((o1%6)*6 + o2/6)*96 + (o3%6)*16 + oc
//   m  = (o1/6)*60 + (o2%6)*10 + (o3/6)
// =====================================================================
__global__ void c2s_kernel(const float* __restrict__ O, const float* __restrict__ XS,
                           const float* __restrict__ gamma, float* __restrict__ fm) {
  int idx = blockIdx.x * blockDim.x + threadIdx.x;
  if (idx >= 16 * VOL) return;
  int oc = idx / VOL, pos = idx % VOL;
  int o1 = pos / 2160, r2 = pos % 2160, o2 = r2 / 60, o3 = r2 % 60;
  int Cy = ((o1 % 6) * 6 + o2 / 6) * 96 + (o3 % 6) * 16 + oc;
  int m  = (o1 / 6) * 60 + (o2 % 6) * 10 + (o3 / 6);
  float v = gamma[0] * O[(size_t)Cy * MPAD + m] + XS[(size_t)Cy * MRAW + m];
  fm[idx] = fmaxf(v, 0.f);
}

// =====================================================================
// Deformable conv: trilinear gather on x (zero pad=1), 27 taps, 16->16,
// + bias, ReLU. Weights (27 KB) staged in LDS. Gather-heavy -> VALU.
// =====================================================================
__global__ void deform_kernel(const float* __restrict__ x, const float* __restrict__ off,
                              const float* __restrict__ dw, const float* __restrict__ db,
                              float* __restrict__ out) {
  __shared__ float wsm[16 * 16 * 27];
  for (int i = threadIdx.x; i < 16 * 16 * 27; i += blockDim.x) wsm[i] = dw[i];
  __syncthreads();
  int pos = blockIdx.x * blockDim.x + threadIdx.x;
  if (pos >= VOL) return;
  int z = pos / 2160, rr = pos % 2160, y = rr / 60, xx = rr % 60;
  float acc[16];
  #pragma unroll
  for (int o = 0; o < 16; ++o) acc[o] = db[o];
  for (int n = 0; n < 27; ++n) {
    float pz = (float)(z  + (n / 9))       + off[(size_t)(0*27 + n) * VOL + pos];
    float py = (float)(y  + ((n / 3) % 3)) + off[(size_t)(1*27 + n) * VOL + pos];
    float px = (float)(xx + (n % 3))       + off[(size_t)(2*27 + n) * VOL + pos];
    pz = fminf(fmaxf(pz, 0.f), 61.f);
    py = fminf(fmaxf(py, 0.f), 37.f);
    px = fminf(fmaxf(px, 0.f), 61.f);
    float q0z = fminf(floorf(pz), 60.f);
    float q0y = fminf(floorf(py), 36.f);
    float q0x = fminf(floorf(px), 60.f);
    float fz = pz - q0z, fy = py - q0y, fx = px - q0x;
    int z0 = (int)q0z, y0 = (int)q0y, x0 = (int)q0x;
    float val[16];
    #pragma unroll
    for (int c = 0; c < 16; ++c) val[c] = 0.f;
    #pragma unroll
    for (int dz = 0; dz < 2; ++dz) {
      float wz = dz ? fz : 1.f - fz;
      int zq = z0 + dz - 1;
      if (zq < 0 || zq >= 60) continue;
      #pragma unroll
      for (int dy = 0; dy < 2; ++dy) {
        float wy = dy ? fy : 1.f - fy;
        int yq = y0 + dy - 1;
        if (yq < 0 || yq >= 36) continue;
        #pragma unroll
        for (int dx = 0; dx < 2; ++dx) {
          float wx = dx ? fx : 1.f - fx;
          int xq = x0 + dx - 1;
          if (xq < 0 || xq >= 60) continue;
          float wgt = wz * wy * wx;
          const float* bp = x + (size_t)(zq * 36 + yq) * 60 + xq;
          #pragma unroll
          for (int c = 0; c < 16; ++c) val[c] += wgt * bp[(size_t)c * VOL];
        }
      }
    }
    #pragma unroll
    for (int o = 0; o < 16; ++o) {
      float s = 0.f;
      const float* wrow = &wsm[(o * 16) * 27 + n];
      #pragma unroll
      for (int c = 0; c < 16; ++c) s += wrow[c * 27] * val[c];
      acc[o] += s;
    }
  }
  #pragma unroll
  for (int o = 0; o < 16; ++o) out[(size_t)o * VOL + pos] = fmaxf(acc[o], 0.f);
}

// =====================================================================
// Host launcher
// =====================================================================
extern "C" void kernel_launch(void* const* d_in, const int* in_sizes, int n_in,
                              void* d_out, int out_size, void* d_ws, size_t ws_size,
                              hipStream_t stream) {
  const float* x      = (const float*)d_in[0];
  const float* ref_w  = (const float*)d_in[1];
  const float* ref_b  = (const float*)d_in[2];
  const float* oc1_w  = (const float*)d_in[3];
  const float* oc1_b  = (const float*)d_in[4];
  const float* gamma1 = (const float*)d_in[5];
  const float* oc2_w  = (const float*)d_in[6];
  const float* oc2_b  = (const float*)d_in[7];
  const float* gamma2 = (const float*)d_in[8];
  const float* oc3_w  = (const float*)d_in[9];
  const float* oc3_b  = (const float*)d_in[10];
  const float* dw     = (const float*)d_in[11];
  const float* db     = (const float*)d_in[12];
  float* out = (float*)d_out;

  // workspace layout (bytes); E (48 MB) is aliased as the 81-ch offset buffer (42 MB)
  char* base = (char*)d_ws;
  float*  bufA = (float*)(base);                  //  8,294,400  feature map A
  float*  bufB = (float*)(base + 8294400);        //  8,294,400  feature map B
  float*  XS   = (float*)(base + 16588800);       //  8,294,400  f32 Q [3456][600]
  __bf16* QB   = (__bf16*)(base + 24883200);      //  4,202,496  bf16 Q [3456][608]
  __bf16* QT   = (__bf16*)(base + 29085696);      //  4,202,496  bf16 Q^T [608][3456]
  float*  E    = (float*)(base + 33288192);       // 47,775,744  energy/attn (alias: offsets)
  float*  O    = (float*)(base + 81063936);       //  8,404,992  attn output [3456][608]
  __bf16* Wb1  = (__bf16*)(base + 89468928);      //     55,296  conv1 weights 16x1728
  __bf16* WbA  = (__bf16*)(base + 89524224);      //     14,336  oc1 weights 16x448
  __bf16* WbB  = (__bf16*)(base + 89538560);      //     14,336  oc2 weights 16x448
  __bf16* Wb3  = (__bf16*)(base + 89552896);      //     86,016  oc3 weights 96x448

  dim3 b256(256);
  dim3 convGrid(2160, 4);      // (z*36+y) x {x0 = 0,16,32,44}
  dim3 convBlk(32);            // one wave per workgroup

  // weight repacks (bf16, tap-major, zero-padded)
  repack_w<<<(16*KC1 + 255)/256, b256, 0, stream>>>(ref_w, Wb1, 16, 64, 27, 16*KC1);
  repack_w<<<(16*KC16 + 255)/256, b256, 0, stream>>>(oc1_w, WbA, 16, 16, 28, 16*KC16);
  repack_w<<<(16*KC16 + 255)/256, b256, 0, stream>>>(oc2_w, WbB, 16, 16, 28, 16*KC16);
  repack_w<<<(96*KC16 + 255)/256, b256, 0, stream>>>(oc3_w, Wb3, 81, 16, 28, 96*KC16);

  conv1_wmma<<<convGrid, convBlk, 0, stream>>>(x, Wb1, ref_b, bufA);

  auto attn_block = [&](const __bf16* Wb, const float* cb, const float* gamma) {
    conv_wmma_c16<1, false><<<convGrid, convBlk, 0, stream>>>(bufA, Wb, cb, bufB, 16);
    s2c_kernel<<<(CP * MPAD + 255) / 256, b256, 0, stream>>>(bufB, XS, QB, QT);
    energy_wmma<<<dim3(CP / 16, CP / 128), dim3(64), 0, stream>>>(QB, E);
    softmax_row<<<CP, b256, 0, stream>>>(E);
    outmm_wmma<<<dim3(CP / 16, MPAD / 32), dim3(32), 0, stream>>>(E, QT, O);
    c2s_kernel<<<(16 * VOL + 255) / 256, b256, 0, stream>>>(O, XS, gamma, bufA);
  };
  attn_block(WbA, oc1_b, gamma1);
  attn_block(WbB, oc2_b, gamma2);

  conv_wmma_c16<6, false><<<convGrid, convBlk, 0, stream>>>(bufA, Wb3, oc3_b, E, 81);
  deform_kernel<<<(VOL + 127) / 128, dim3(128), 0, stream>>>(x, E, dw, db, out);
}
